// SpatialCrossAttention_11665131175990
// MI455X (gfx1250) — matrix operand order repaired and checked by reading, not compiled, not run on hardware
//
#include <hip/hip_runtime.h>
#include <hip/hip_bf16.h>

// ---------------- problem constants ----------------
#define BATCH   16
#define CCH     320          // channels / query dim / inner dim
#define SP      4096         // H*W tokens
#define HEADS   8
#define HDIM    40
#define LCTX    77
#define CTXD    768
#define NGRP    32
#define CPG     (CCH / NGRP) // 10
#define MKV_PAD 1280         // B*LCTX = 1232 padded to 5 full 256-row GEMM blocks

typedef __attribute__((ext_vector_type(16))) _Float16 v16h;
typedef __attribute__((ext_vector_type(8)))  float    v8f;

union HV {
    uint4 q[2];
    v16h  h;
};

static __device__ __forceinline__ v8f wmma_f16(v16h a, v16h b, v8f c) {
    return __builtin_amdgcn_wmma_f32_16x16x32_f16(false, a, false, b, (short)0, c,
                                                  false, false);
}
// ISA 7.12.2: 16-bit A 16x32 — lane L holds row M=L&15; K = e + (e>=8?8:0) + ((L&16)?8:0)
static __device__ __forceinline__ int a_frag_k(int e, int lane) {
    return e + (e >= 8 ? 8 : 0) + ((lane & 16) ? 8 : 0);
}
// B 32x16 — lane L holds col N=L&15; K = e + (e>=8?8:0) + ((L&16)?8:0)
static __device__ __forceinline__ int b_frag_k(int e, int lane) {
    return e + (e >= 8 ? 8 : 0) + ((lane & 16) ? 8 : 0);
}

// ---------------- GroupNorm stats (wave-shuffle reduce, 1 LDS slot per wave) ---------
__global__ __launch_bounds__(256) void gn_stats_kernel(const float* __restrict__ x,
                                                       float* __restrict__ stats) {
    __shared__ float wsum[8], wsumsq[8];
    const int bg = blockIdx.x;                       // b*NGRP + g
    const float* base = x + (size_t)bg * CPG * SP;   // contiguous 10*4096 floats
    float s = 0.f, ss = 0.f;
    for (int i = threadIdx.x; i < CPG * SP; i += 256) {
        float v = base[i];
        s += v; ss += v * v;
    }
    #pragma unroll
    for (int o = 16; o >= 1; o >>= 1) {
        s  += __shfl_xor(s,  o, 32);
        ss += __shfl_xor(ss, o, 32);
    }
    const int lane = threadIdx.x & 31, wave = threadIdx.x >> 5;
    if (lane == 0) { wsum[wave] = s; wsumsq[wave] = ss; }
    __syncthreads();
    if (threadIdx.x < 8) {
        s = wsum[threadIdx.x]; ss = wsumsq[threadIdx.x];
        #pragma unroll
        for (int o = 4; o >= 1; o >>= 1) {
            s  += __shfl_xor(s,  o, 32);
            ss += __shfl_xor(ss, o, 32);
        }
        if (threadIdx.x == 0) {
            const float inv_n = 1.0f / (CPG * SP);
            float mean = s * inv_n;
            float var  = ss * inv_n - mean * mean;
            stats[bg * 2 + 0] = mean;
            stats[bg * 2 + 1] = rsqrtf(var + 1e-5f);
        }
    }
}

// ---- GroupNorm apply + [B,C,S] -> [B,S,C] transpose through LDS, fp16 tokens ----
// grid: (S/32, C/32, B); block: 32x8
__global__ __launch_bounds__(256) void gn_tokens_kernel(const float* __restrict__ x,
                                                        const float* __restrict__ stats,
                                                        const float* __restrict__ gscale,
                                                        const float* __restrict__ gbias,
                                                        _Float16* __restrict__ tokens) {
    __shared__ float tile[32][33];
    const int tx = threadIdx.x, ty = threadIdx.y;
    const int sb = blockIdx.x * 32, cb = blockIdx.y * 32, b = blockIdx.z;
    #pragma unroll
    for (int r = 0; r < 4; ++r) {
        int c = cb + ty + r * 8;
        int s = sb + tx;                              // coalesced read over s
        float mean = stats[(b * NGRP + c / CPG) * 2 + 0];
        float rstd = stats[(b * NGRP + c / CPG) * 2 + 1];
        float v = x[((size_t)b * CCH + c) * SP + s];
        tile[ty + r * 8][tx] = (v - mean) * rstd * gscale[c] + gbias[c];
    }
    __syncthreads();
    #pragma unroll
    for (int r = 0; r < 4; ++r) {
        int s = sb + ty + r * 8;
        int c = cb + tx;                              // coalesced write over c
        tokens[((size_t)b * SP + s) * CCH + c] = (_Float16)tile[tx][ty + r * 8];
    }
}

// ---------------- fp32 -> fp16 copy (context) ----------------
__global__ __launch_bounds__(256) void f32_to_f16_kernel(const float* __restrict__ src,
                                                         _Float16* __restrict__ dst,
                                                         size_t n) {
    size_t idx = (size_t)blockIdx.x * 256 + threadIdx.x;
    if (idx < n) dst[idx] = (_Float16)src[idx];
}

// -------- weight -> WMMA-B-fragment pack: W[outF=N, inF=K] -> frag[kT][nT][lane][e] ----
__global__ __launch_bounds__(256) void wt_frag_kernel(const float* __restrict__ W,
                                                      _Float16* __restrict__ dst,
                                                      int N, int K) {
    size_t idx = (size_t)blockIdx.x * 256 + threadIdx.x;
    if (idx >= (size_t)K * N) return;
    int e    = (int)(idx & 15);
    int lane = (int)((idx >> 4) & 31);
    int frag = (int)(idx >> 9);
    int nT   = frag % (N >> 4);
    int kT   = frag / (N >> 4);
    int k = kT * 32 + b_frag_k(e, lane);
    int n = nT * 16 + (lane & 15);
    dst[idx] = (_Float16)W[(size_t)n * K + k];        // coalesced fragment-order writes
}

// -------- K-fragment pack for scores: [b][h][ks:2][lt:5][lane][e], zero-padded --------
__global__ __launch_bounds__(256) void kfrag_kernel(const _Float16* __restrict__ Kb,
                                                    _Float16* __restrict__ dst) {
    size_t idx = (size_t)blockIdx.x * 256 + threadIdx.x;
    if (idx >= (size_t)BATCH * HEADS * 2 * 5 * 512) return;
    int e    = (int)(idx & 15);
    int lane = (int)((idx >> 4) & 31);
    int lt   = (int)((idx >> 9) % 5);
    int ks   = (int)((idx / (512 * 5)) % 2);
    int h    = (int)((idx / (512 * 5 * 2)) % HEADS);
    int b    = (int)(idx / ((size_t)512 * 5 * 2 * HEADS));
    int d = ks * 32 + b_frag_k(e, lane);              // K-dim = head dim
    int l = lt * 16 + (lane & 15);                    // N-dim = context pos
    _Float16 v = (_Float16)0.f;
    if (d < HDIM && l < LCTX)
        v = Kb[((size_t)b * LCTX + l) * CCH + h * HDIM + d];
    dst[idx] = v;
}

// -------- V-fragment pack for P@V: [b][h][ks:3][nt:3][lane][e], zero-padded ----------
__global__ __launch_bounds__(256) void vfrag_kernel(const _Float16* __restrict__ Vb,
                                                    _Float16* __restrict__ dst) {
    size_t idx = (size_t)blockIdx.x * 256 + threadIdx.x;
    if (idx >= (size_t)BATCH * HEADS * 3 * 3 * 512) return;
    int e    = (int)(idx & 15);
    int lane = (int)((idx >> 4) & 31);
    int nt   = (int)((idx >> 9) % 3);
    int ks   = (int)((idx / (512 * 3)) % 3);
    int h    = (int)((idx / (512 * 3 * 3)) % HEADS);
    int b    = (int)(idx / ((size_t)512 * 3 * 3 * HEADS));
    int l = ks * 32 + b_frag_k(e, lane);              // K-dim = context pos
    int d = nt * 16 + (lane & 15);                    // N-dim = head dim
    _Float16 v = (_Float16)0.f;
    if (l < LCTX && d < HDIM)
        v = Vb[((size_t)b * LCTX + l) * CCH + h * HDIM + d];
    dst[idx] = v;
}

// ---------------- register-resident f16 GEMM: C[M,N] = A[M,K] * Bfrag ---------------
// Guard-free (all M padded to multiples of 256; K multiple of 64), software-pipelined
// over two fragment register sets per 64-wide K step. No LDS, no barriers.
// Block: 256 thr = 8 waves; block tile 256M x 64N; wave tile 64M x 32N.
__global__ __launch_bounds__(256) void gemm_f16_kernel(const _Float16* __restrict__ A,
                                                       const _Float16* __restrict__ Bfrag,
                                                       _Float16* __restrict__ Cout,
                                                       int N, int K) {
    const int t = threadIdx.x;
    const int lane = t & 31;
    const int wave = t >> 5;
    const int wm = (wave & 3) * 64;
    const int wn = (wave >> 2) * 32;
    const int blockM = blockIdx.y * 256;
    const int blockN = blockIdx.x * 64;
    const int halfSel = lane >> 4;
    const int rowBase = blockM + wm + (lane & 15);
    const int nFrag16 = N >> 4;
    const int fragN0 = (blockN >> 4) + (wave >> 2) * 2;

    v8f acc[4][2];
    const v8f vz = {0.f, 0.f, 0.f, 0.f, 0.f, 0.f, 0.f, 0.f};
    #pragma unroll
    for (int mi = 0; mi < 4; ++mi) { acc[mi][0] = vz; acc[mi][1] = vz; }

    auto loadB = [&](int kk, v16h bf[2]) {
        #pragma unroll
        for (int ni = 0; ni < 2; ++ni) {
            const _Float16* bp =
                Bfrag + ((size_t)(kk >> 5) * nFrag16 + fragN0 + ni) * 512 + lane * 16;
            HV tmp;
            tmp.q[0] = *(const uint4*)(bp);
            tmp.q[1] = *(const uint4*)(bp + 8);
            bf[ni] = tmp.h;
        }
    };
    auto loadA = [&](int kk, v16h af[4]) {
        #pragma unroll
        for (int mi = 0; mi < 4; ++mi) {
            const _Float16* ap = A + (size_t)(rowBase + mi * 16) * K + kk + halfSel * 8;
            HV a;
            a.q[0] = *(const uint4*)(ap);          // K run {0..7}+8*half
            a.q[1] = *(const uint4*)(ap + 16);     // K run {16..23}+8*half
            af[mi] = a.h;
        }
    };

    v16h afA[4], bfA[2], afB[4], bfB[2];
    loadA(0, afA);
    loadB(0, bfA);
    for (int k0 = 0; k0 < K; k0 += 64) {
        loadA(k0 + 32, afB);
        loadB(k0 + 32, bfB);
        {   // speculative prefetch of the K-slab two steps ahead (clamped address)
            int kp = (k0 + 128 < K) ? k0 + 128 : 0;
            __builtin_prefetch(A + (size_t)rowBase * K + kp, 0, 3);
        }
        #pragma unroll
        for (int mi = 0; mi < 4; ++mi)
            #pragma unroll
            for (int ni = 0; ni < 2; ++ni)
                acc[mi][ni] = wmma_f16(afA[mi], bfA[ni], acc[mi][ni]);
        int kn = (k0 + 64 < K) ? k0 + 64 : 0;      // last step: harmless reload
        loadA(kn, afA);
        loadB(kn, bfA);
        #pragma unroll
        for (int mi = 0; mi < 4; ++mi)
            #pragma unroll
            for (int ni = 0; ni < 2; ++ni)
                acc[mi][ni] = wmma_f16(afB[mi], bfB[ni], acc[mi][ni]);
    }
    // C layout: row = e + 8*(lane>=16), col = lane&15
    const int rh = halfSel * 8;
    #pragma unroll
    for (int mi = 0; mi < 4; ++mi)
        #pragma unroll
        for (int ni = 0; ni < 2; ++ni)
            #pragma unroll
            for (int e = 0; e < 8; ++e) {
                int row = blockM + wm + mi * 16 + e + rh;
                int col = blockN + wn + ni * 16 + (lane & 15);
                Cout[(size_t)row * N + col] = (_Float16)acc[mi][ni][e];
            }
}

// ---------------- fused attention: 4 waves/block, one 16-query tile per wave ----------
#define P_STRIDE 104
__global__ __launch_bounds__(128) void attn_kernel(const _Float16* __restrict__ Q,
                                                   const _Float16* __restrict__ Kf,
                                                   const _Float16* __restrict__ Vf,
                                                   _Float16* __restrict__ AO) {
    __shared__ _Float16 Pt[4][16 * P_STRIDE];
    const int lane = threadIdx.x & 31;
    const int wid  = threadIdx.x >> 5;
    const int tile = blockIdx.x * 4 + wid;
    const int sT = tile & 255;
    const int h  = (tile >> 8) & 7;
    const int b  = tile >> 11;
    const int s0 = sT * 16;
    const float scale = 0.158113883f;                 // 1/sqrt(40)
    const int row = lane & 15;
    const int halfSel = lane >> 4;
    const uint4 z4 = make_uint4(0u, 0u, 0u, 0u);

    // Q fragments: frag0 fully in-bounds (d in {0..31}); frag1 only lanes<16,e<8 (d 32..39)
    const _Float16* qp = Q + ((size_t)b * SP + s0 + row) * CCH + h * HDIM;
    HV q0, q1;
    q0.q[0] = *(const uint4*)(qp + halfSel * 8);
    q0.q[1] = *(const uint4*)(qp + 16 + halfSel * 8);
    uint4 t32 = *(const uint4*)(qp + 32);             // unconditional; select-zero below
    q1.q[0] = halfSel ? z4 : t32;                     // v_cndmask, no exec branch
    q1.q[1] = z4;

    // scores: 16 x 80 (Lc padded), 5 N-tiles x 2 K-steps, fragment loads are 32B/lane
    const v8f vz = {0.f, 0.f, 0.f, 0.f, 0.f, 0.f, 0.f, 0.f};
    v8f sc[5];
    const _Float16* kfb = Kf + ((size_t)(b * HEADS + h) * 2) * 5 * 512 + lane * 16;
    #pragma unroll
    for (int tN = 0; tN < 5; ++tN) {
        HV k0f, k1f;
        k0f.q[0] = *(const uint4*)(kfb + (size_t)tN * 512);
        k0f.q[1] = *(const uint4*)(kfb + (size_t)tN * 512 + 8);
        k1f.q[0] = *(const uint4*)(kfb + (size_t)(5 + tN) * 512);
        k1f.q[1] = *(const uint4*)(kfb + (size_t)(5 + tN) * 512 + 8);
        v8f a = wmma_f16(q0.h, k0f.h, vz);
        sc[tN] = wmma_f16(q1.h, k1f.h, a);
    }

    // softmax over 80 cols (mask >=77); lanes<16 own rows 0..7 (e), lanes>=16 rows 8..15
    const int col0 = lane & 15;
    #pragma unroll
    for (int e = 0; e < 8; ++e) {
        #pragma unroll
        for (int tN = 0; tN < 5; ++tN) {
            int col = col0 + 16 * tN;
            float v = sc[tN][e];
            sc[tN][e] = (col < LCTX) ? v * scale : -1e30f;
        }
        float m = sc[0][e];
        #pragma unroll
        for (int tN = 1; tN < 5; ++tN) m = fmaxf(m, sc[tN][e]);
        m = fmaxf(m, __shfl_xor(m, 1, 32));
        m = fmaxf(m, __shfl_xor(m, 2, 32));
        m = fmaxf(m, __shfl_xor(m, 4, 32));
        m = fmaxf(m, __shfl_xor(m, 8, 32));
        float sum = 0.f;
        #pragma unroll
        for (int tN = 0; tN < 5; ++tN) {
            float p = __expf(sc[tN][e] - m);
            sc[tN][e] = p;
            sum += p;
        }
        sum += __shfl_xor(sum, 1, 32);
        sum += __shfl_xor(sum, 2, 32);
        sum += __shfl_xor(sum, 4, 32);
        sum += __shfl_xor(sum, 8, 32);
        float inv = 1.0f / sum;
        #pragma unroll
        for (int tN = 0; tN < 5; ++tN) sc[tN][e] *= inv;
    }

    // C-layout -> A-layout through LDS (pad K 80 -> 96 with zeros)
    const int rh = halfSel * 8;
    _Float16* myP = Pt[wid];
    for (int i = lane; i < 16 * 16; i += 32)
        myP[(i >> 4) * P_STRIDE + 80 + (i & 15)] = (_Float16)0.f;
    #pragma unroll
    for (int tN = 0; tN < 5; ++tN)
        #pragma unroll
        for (int e = 0; e < 8; ++e)
            myP[(rh + e) * P_STRIDE + col0 + 16 * tN] = (_Float16)sc[tN][e];
    __syncthreads();

    v16h pf[3];
    #pragma unroll
    for (int ks = 0; ks < 3; ++ks)
        #pragma unroll
        for (int e = 0; e < 16; ++e)
            pf[ks][e] = myP[row * P_STRIDE + ks * 32 + a_frag_k(e, lane)];

    // out = P(16x96) @ V(96x48-padded); V fragments contiguous 32B/lane
    const _Float16* vfb = Vf + ((size_t)(b * HEADS + h) * 9) * 512 + lane * 16;
    v8f oa[3];
    #pragma unroll
    for (int nt = 0; nt < 3; ++nt) oa[nt] = vz;
    #pragma unroll
    for (int ks = 0; ks < 3; ++ks)
        #pragma unroll
        for (int nt = 0; nt < 3; ++nt) {
            HV vf;
            vf.q[0] = *(const uint4*)(vfb + (size_t)(ks * 3 + nt) * 512);
            vf.q[1] = *(const uint4*)(vfb + (size_t)(ks * 3 + nt) * 512 + 8);
            oa[nt] = wmma_f16(pf[ks], vf.h, oa[nt]);
        }
    // store: nt 0,1 always in-bounds (d<32); nt 2 guarded (d = 32..47)
    const size_t aoBase = ((size_t)b * SP + s0 + rh) * CCH + h * HDIM;
    #pragma unroll
    for (int nt = 0; nt < 2; ++nt)
        #pragma unroll
        for (int e = 0; e < 8; ++e)
            AO[aoBase + (size_t)e * CCH + nt * 16 + col0] = (_Float16)oa[nt][e];
    if (col0 < 8) {
        #pragma unroll
        for (int e = 0; e < 8; ++e)
            AO[aoBase + (size_t)e * CCH + 32 + col0] = (_Float16)oa[2][e];
    }
}

// ---- fused bias + residual + [B,S,C] -> [B,C,S] transpose (coalesced both sides) ----
// grid: (S/32, C/32, B); block 32x8
__global__ __launch_bounds__(256) void bias_res_tr_kernel(const _Float16* __restrict__ O,
                                                          const float* __restrict__ bo,
                                                          const float* __restrict__ xres,
                                                          float* __restrict__ outp) {
    __shared__ float tile[32][33];
    const int tx = threadIdx.x, ty = threadIdx.y;
    const int sb = blockIdx.x * 32, cb = blockIdx.y * 32, b = blockIdx.z;
    #pragma unroll
    for (int r = 0; r < 4; ++r) {
        int s = sb + ty + r * 8;
        int c = cb + tx;                              // coalesced read over c
        tile[tx][ty + r * 8] = (float)O[((size_t)b * SP + s) * CCH + c];
    }
    __syncthreads();
    #pragma unroll
    for (int r = 0; r < 4; ++r) {
        int c = cb + ty + r * 8;
        int s = sb + tx;                              // coalesced write over s
        size_t oidx = ((size_t)b * CCH + c) * SP + s;
        outp[oidx] = tile[ty + r * 8][tx] + bo[c] + xres[oidx];
    }
}

// ---------------- host-side orchestration ----------------
extern "C" void kernel_launch(void* const* d_in, const int* in_sizes, int n_in,
                              void* d_out, int out_size, void* d_ws, size_t ws_size,
                              hipStream_t stream) {
    const float* x      = (const float*)d_in[0];
    const float* ctx    = (const float*)d_in[1];
    const float* Wq     = (const float*)d_in[2];
    const float* Wk     = (const float*)d_in[3];
    const float* Wv     = (const float*)d_in[4];
    const float* Wo     = (const float*)d_in[5];
    const float* bo     = (const float*)d_in[6];
    const float* gscale = (const float*)d_in[7];
    const float* gbias  = (const float*)d_in[8];
    float* outp = (float*)d_out;

    char* ws = (char*)d_ws;
    size_t off = 0;
    auto take = [&](size_t bytes) {
        char* p = ws + off;
        off = (off + bytes + 255) & ~(size_t)255;
        return p;
    };
    const size_t tokN   = (size_t)BATCH * SP * CCH;        // 20,971,520 halves
    const size_t kvNp   = (size_t)MKV_PAD * CCH;           // padded K/V rows
    const size_t ctxNv  = (size_t)BATCH * LCTX * CTXD;     // valid context elements
    const size_t ctxNp  = (size_t)MKV_PAD * CTXD;          // padded context rows
    const size_t kfN    = (size_t)BATCH * HEADS * 2 * 5 * 512;
    const size_t vfN    = (size_t)BATCH * HEADS * 3 * 3 * 512;

    float*    stats  = (float*)   take(BATCH * NGRP * 2 * sizeof(float));
    _Float16* tokens = (_Float16*)take(tokN * 2);
    _Float16* Qb     = (_Float16*)take(tokN * 2);
    _Float16* Kb     = (_Float16*)take(kvNp * 2);
    _Float16* Vb     = (_Float16*)take(kvNp * 2);
    _Float16* AOb    = (_Float16*)take(tokN * 2);
    _Float16* Ob     = (_Float16*)take(tokN * 2);
    _Float16* ctxH   = (_Float16*)take(ctxNp * 2);
    _Float16* WqF    = (_Float16*)take((size_t)CCH * CCH * 2);
    _Float16* WkF    = (_Float16*)take((size_t)CTXD * CCH * 2);
    _Float16* WvF    = (_Float16*)take((size_t)CTXD * CCH * 2);
    _Float16* WoF    = (_Float16*)take((size_t)CCH * CCH * 2);
    _Float16* KfB    = (_Float16*)take(kfN * 2);
    _Float16* VfB    = (_Float16*)take(vfN * 2);
    (void)ws_size;

    // 1) GroupNorm -> fp16 tokens in [B,S,C] (LDS tile transpose, coalesced)
    gn_stats_kernel<<<BATCH * NGRP, 256, 0, stream>>>(x, stats);
    gn_tokens_kernel<<<dim3(SP / 32, CCH / 32, BATCH), dim3(32, 8), 0, stream>>>(
        x, stats, gscale, gbias, tokens);

    // 2) fp16 conversions: context + fragment-packed weights
    f32_to_f16_kernel<<<(int)((ctxNv + 255) / 256), 256, 0, stream>>>(ctx, ctxH, ctxNv);
    wt_frag_kernel<<<(CCH * CCH + 255) / 256, 256, 0, stream>>>(Wq, WqF, CCH, CCH);
    wt_frag_kernel<<<(CCH * CTXD + 255) / 256, 256, 0, stream>>>(Wk, WkF, CCH, CTXD);
    wt_frag_kernel<<<(CCH * CTXD + 255) / 256, 256, 0, stream>>>(Wv, WvF, CCH, CTXD);
    wt_frag_kernel<<<(CCH * CCH + 255) / 256, 256, 0, stream>>>(Wo, WoF, CCH, CCH);

    // 3) projections — guard-free pipelined WMMA GEMMs (M padded to 256 multiples;
    //    K/V rows 1232..1279 are garbage but never consumed downstream)
    const int Mq = BATCH * SP;      // 65536
    gemm_f16_kernel<<<dim3(CCH / 64, Mq / 256), 256, 0, stream>>>(
        tokens, WqF, Qb, CCH, CCH);
    gemm_f16_kernel<<<dim3(CCH / 64, MKV_PAD / 256), 256, 0, stream>>>(
        ctxH, WkF, Kb, CCH, CTXD);
    gemm_f16_kernel<<<dim3(CCH / 64, MKV_PAD / 256), 256, 0, stream>>>(
        ctxH, WvF, Vb, CCH, CTXD);

    // 4) K/V fragment repack (zero-padding baked in)
    kfrag_kernel<<<(int)((kfN + 255) / 256), 256, 0, stream>>>(Kb, KfB);
    vfrag_kernel<<<(int)((vfN + 255) / 256), 256, 0, stream>>>(Vb, VfB);

    // 5) fused attention
    attn_kernel<<<BATCH * HEADS * (SP / 16) / 4, 128, 0, stream>>>(Qb, KfB, VfB, AOb);

    // 6) out-projection (row-major fp16, coalesced) + fused bias/residual/transpose
    gemm_f16_kernel<<<dim3(CCH / 64, Mq / 256), 256, 0, stream>>>(
        AOb, WoF, Ob, CCH, CCH);
    bias_res_tr_kernel<<<dim3(SP / 32, CCH / 32, BATCH), dim3(32, 8), 0, stream>>>(
        Ob, bo, x, outp);
    (void)in_sizes; (void)n_in; (void)out_size;
}
